// CustomExpert_13855564497414
// MI455X (gfx1250) — compile-verified, matrix operand order checked by reading, not run on hardware
//
#include <hip/hip_runtime.h>

typedef __attribute__((ext_vector_type(16))) __bf16 v16bf;
typedef __attribute__((ext_vector_type(8)))  __bf16 v8bf;
typedef __attribute__((ext_vector_type(2)))  __bf16 v2bf;
typedef __attribute__((ext_vector_type(2)))  float  v2f;
typedef __attribute__((ext_vector_type(8)))  float  v8f;
typedef unsigned short ushort_t;

#define E_ 8
#define T_ 2048
#define D_ 1024
#define H_ 4096

#define BK 32
#define LDT 40   // padded LDS row stride in bf16 elements (80 bytes)

__device__ __forceinline__ ushort_t f2bf(float f) {
  __bf16 b = (__bf16)f;
  return __builtin_bit_cast(ushort_t, b);
}
__device__ __forceinline__ unsigned cvt2(float x, float y) {
  v2f f; f.x = x; f.y = y;
  v2bf b = __builtin_convertvector(f, v2bf);
  return __builtin_bit_cast(unsigned, b);
}

// One 16x32 bf16 operand fragment from an LDS tile stored row-major
// (row = M for A, row = N for B^T), stride LDT.
// Lane<16 holds K [0..7]+[16..23]; lane>=16 holds K [8..15]+[24..31].
__device__ __forceinline__ v16bf frag_load(const ushort_t* base, int row, int half) {
  const v8bf lo = *reinterpret_cast<const v8bf*>(base + row * LDT + half);
  const v8bf hi = *reinterpret_cast<const v8bf*>(base + row * LDT + 16 + half);
  return __builtin_shufflevector(lo, hi, 0,1,2,3,4,5,6,7,8,9,10,11,12,13,14,15);
}

// 64x64 wave tile: 4x4 fragments, 16 WMMAs per 8 fragment loads.
__device__ __forceinline__ void mma_tile4(const ushort_t* As, const ushort_t* Bs,
                                          int aRow0, int bRow0, int lr, int half,
                                          v8f (&acc)[4][4]) {
  v16bf a[4];
#pragma unroll
  for (int fm = 0; fm < 4; ++fm) a[fm] = frag_load(As, aRow0 + fm * 16 + lr, half);
#pragma unroll
  for (int fn = 0; fn < 4; ++fn) {
    const v16bf b = frag_load(Bs, bRow0 + fn * 16 + lr, half);
#pragma unroll
    for (int fm = 0; fm < 4; ++fm)
      acc[fm][fn] = __builtin_amdgcn_wmma_f32_16x16x32_bf16(
          false, a[fm], false, b, (short)0, acc[fm][fn], false, false);
  }
}

// ---------------------------------------------------------------------------
// Kernel 1: Y = relu(X @ W1 + b1). X fp32 [T,D], W1 fp32 [D,H], Y bf16 [T,H].
// Block tile 128x256 (2x4 waves of 64x64).
// ---------------------------------------------------------------------------
#define K1_BM 128
#define K1_BN 256
__global__ __launch_bounds__(256) void moe_fc1_relu(
    const float* __restrict__ X, const float* __restrict__ W1,
    const float* __restrict__ B1, ushort_t* __restrict__ Y) {
  __shared__ __align__(16) ushort_t As[2][K1_BM * LDT];
  __shared__ __align__(16) ushort_t Bs[2][K1_BN * LDT];

  const int tid = threadIdx.x;
  const int e = blockIdx.z, tm = blockIdx.y, tn = blockIdx.x;

  const float* Ag = X + (size_t)e * T_ * D_ + (size_t)tm * K1_BM * D_;
  const float* Bg = W1 + (size_t)e * D_ * H_ + (size_t)tn * K1_BN;
  const float* biasg = B1 + (size_t)e * H_ + (size_t)tn * K1_BN;
  ushort_t* Yg = Y + (size_t)e * T_ * H_ + (size_t)tm * K1_BM * H_ + (size_t)tn * K1_BN;

  const int wave = tid >> 5;
  const int lane = tid & 31;
  const int wm = wave & 1;        // 2 row groups of 64
  const int wn = wave >> 1;       // 4 col groups of 64
  const int lr = lane & 15;
  const int half = (lane & 16) ? 8 : 0;

  // A: 128x32 floats, row-contiguous float4 loads (16 floats/thread)
  const int arow = tid >> 3;            // + j*32
  const int ac4 = (tid & 7) << 2;
  // B: 32x256 floats fetched as float2 column-pairs (32 floats/thread),
  // thread owns columns {bnp, bnp+1}, k-range [bkg, bkg+16)
  const int bnp = (tid & 127) << 1;
  const int bkg = (tid >> 7) << 4;      // 0 or 16

  v8f acc[4][4] = {};

  auto fetchA = [&](int kt, float4 (&ra)[4]) {
#pragma unroll
    for (int j = 0; j < 4; ++j)
      ra[j] = *reinterpret_cast<const float4*>(
          Ag + (size_t)(arow + j * 32) * D_ + kt * BK + ac4);
  };
  auto fetchB = [&](int kt, float2 (&rb)[16]) {
#pragma unroll
    for (int i = 0; i < 16; ++i)
      rb[i] = *reinterpret_cast<const float2*>(
          Bg + (size_t)(kt * BK + bkg + i) * H_ + bnp);
  };
  auto commit = [&](int buf, const float4 (&ra)[4], const float2 (&rb)[16]) {
#pragma unroll
    for (int j = 0; j < 4; ++j) {
      uint2 p; p.x = cvt2(ra[j].x, ra[j].y); p.y = cvt2(ra[j].z, ra[j].w);
      *reinterpret_cast<uint2*>(&As[buf][(arow + j * 32) * LDT + ac4]) = p;
    }
    uint4 u;
    u.x = cvt2(rb[0].x, rb[1].x);  u.y = cvt2(rb[2].x, rb[3].x);
    u.z = cvt2(rb[4].x, rb[5].x);  u.w = cvt2(rb[6].x, rb[7].x);
    *reinterpret_cast<uint4*>(&Bs[buf][bnp * LDT + bkg]) = u;
    u.x = cvt2(rb[8].x, rb[9].x);  u.y = cvt2(rb[10].x, rb[11].x);
    u.z = cvt2(rb[12].x, rb[13].x); u.w = cvt2(rb[14].x, rb[15].x);
    *reinterpret_cast<uint4*>(&Bs[buf][bnp * LDT + bkg + 8]) = u;
    u.x = cvt2(rb[0].y, rb[1].y);  u.y = cvt2(rb[2].y, rb[3].y);
    u.z = cvt2(rb[4].y, rb[5].y);  u.w = cvt2(rb[6].y, rb[7].y);
    *reinterpret_cast<uint4*>(&Bs[buf][(bnp + 1) * LDT + bkg]) = u;
    u.x = cvt2(rb[8].y, rb[9].y);  u.y = cvt2(rb[10].y, rb[11].y);
    u.z = cvt2(rb[12].y, rb[13].y); u.w = cvt2(rb[14].y, rb[15].y);
    *reinterpret_cast<uint4*>(&Bs[buf][(bnp + 1) * LDT + bkg + 8]) = u;
  };

  const int NK = D_ / BK;  // 32
  {
    float4 ra[4]; float2 rb[16];
    fetchA(0, ra); fetchB(0, rb);
    commit(0, ra, rb);
  }
  __syncthreads();

  int buf = 0;
  for (int kt = 0; kt < NK; ++kt) {
    float4 ra[4]; float2 rb[16];
    if (kt + 1 < NK) {
      fetchA(kt + 1, ra);
      fetchB(kt + 1, rb);
      if (kt + 2 < NK) {
        __builtin_prefetch(Ag + (size_t)arow * D_ + (kt + 2) * BK + ac4, 0, 0);
        __builtin_prefetch(Bg + (size_t)((kt + 2) * BK + bkg) * H_ + bnp, 0, 0);
      }
    }
    mma_tile4(As[buf], Bs[buf], wm * 64, wn * 64, lr, half, acc);
    if (kt + 1 < NK) {
      commit(buf ^ 1, ra, rb);
      __syncthreads();
      buf ^= 1;
    }
  }

  // Epilogue: bias + relu + cvt->bf16. C/D layout: M = half+r, N = lr.
#pragma unroll
  for (int fm = 0; fm < 4; ++fm)
#pragma unroll
    for (int fn = 0; fn < 4; ++fn) {
      const int col = wn * 64 + fn * 16 + lr;
      const float bv = biasg[col];
#pragma unroll
      for (int r = 0; r < 8; ++r) {
        const int row = wm * 64 + fm * 16 + half + r;
        float v = acc[fm][fn][r] + bv;
        v = fmaxf(v, 0.0f);
        Yg[(size_t)row * H_ + col] = f2bf(v);
      }
    }
}

// ---------------------------------------------------------------------------
// Kernel 2: Out = Y @ W2 + b2. Y bf16 [T,H], W2 fp32 [H,D], Out fp32 [T,D].
// Block tile 256x128 (4x2 waves of 64x64). A staged by async global->LDS DMA.
// ---------------------------------------------------------------------------
#define K2_BM 256
#define K2_BN 128
__global__ __launch_bounds__(256) void moe_fc2(
    const ushort_t* __restrict__ Yin, const float* __restrict__ W2,
    const float* __restrict__ B2, float* __restrict__ Out) {
  __shared__ __align__(16) ushort_t As[2][K2_BM * LDT];
  __shared__ __align__(16) ushort_t Bs[2][K2_BN * LDT];

  const int tid = threadIdx.x;
  const int e = blockIdx.z, tm = blockIdx.y, tn = blockIdx.x;

  const ushort_t* Ag = Yin + (size_t)e * T_ * H_ + (size_t)tm * K2_BM * H_;
  const float* Bg = W2 + (size_t)e * H_ * D_ + (size_t)tn * K2_BN;
  const float* biasg = B2 + (size_t)e * D_ + (size_t)tn * K2_BN;
  float* Og = Out + (size_t)e * T_ * D_ + (size_t)tm * K2_BM * D_ + (size_t)tn * K2_BN;

  const int wave = tid >> 5;
  const int lane = tid & 31;
  const int wm = wave & 3;        // 4 row groups of 64
  const int wn = wave >> 2;       // 2 col groups of 64
  const int lr = lane & 15;
  const int half = (lane & 16) ? 8 : 0;

  // A: 256 rows x 32 bf16 = 1024 chunks of 8 bf16; 4 async chunks/thread
  const int arow = tid >> 2;            // + j*64
  const int ac8 = (tid & 3) << 3;
  // B: 32x128 floats, column-strided (16 floats/thread)
  const int bn = tid & 127;
  const int bkg = (tid >> 7) << 4;

  v8f acc[4][4] = {};

  auto asyncA = [&](int buf, int kt) {
#pragma unroll
    for (int j = 0; j < 4; ++j) {
      unsigned lds_off =
          (unsigned)(size_t)(void*)&As[buf][(arow + j * 64) * LDT + ac8];
      unsigned long long gaddr = (unsigned long long)(size_t)(
          Ag + (size_t)(arow + j * 64) * H_ + kt * BK + ac8);
      asm volatile("global_load_async_to_lds_b128 %0, %1, off"
                   :: "v"(lds_off), "v"(gaddr)
                   : "memory");
    }
  };
  auto fetchB = [&](int kt, float (&rb)[16]) {
#pragma unroll
    for (int i = 0; i < 16; ++i)
      rb[i] = Bg[(size_t)(kt * BK + bkg + i) * D_ + bn];
  };
  auto commitB = [&](int buf, const float (&rb)[16]) {
    uint4 u0, u1;
    u0.x = cvt2(rb[0], rb[1]);  u0.y = cvt2(rb[2], rb[3]);
    u0.z = cvt2(rb[4], rb[5]);  u0.w = cvt2(rb[6], rb[7]);
    u1.x = cvt2(rb[8], rb[9]);  u1.y = cvt2(rb[10], rb[11]);
    u1.z = cvt2(rb[12], rb[13]); u1.w = cvt2(rb[14], rb[15]);
    *reinterpret_cast<uint4*>(&Bs[buf][bn * LDT + bkg]) = u0;
    *reinterpret_cast<uint4*>(&Bs[buf][bn * LDT + bkg + 8]) = u1;
  };

  const int NK = H_ / BK;  // 128
  {
    float rb[16];
    asyncA(0, 0);
    fetchB(0, rb);
    commitB(0, rb);
    asm volatile("s_wait_asynccnt 0" ::: "memory");
  }
  __syncthreads();

  int buf = 0;
  for (int kt = 0; kt < NK; ++kt) {
    float rb[16];
    if (kt + 1 < NK) {
      asyncA(buf ^ 1, kt + 1);   // buf^1 free: all waves passed last barrier
      fetchB(kt + 1, rb);
      if (kt + 2 < NK)
        __builtin_prefetch(Bg + (size_t)((kt + 2) * BK + bkg) * D_ + bn, 0, 0);
    }
    mma_tile4(As[buf], Bs[buf], wm * 64, wn * 64, lr, half, acc);
    if (kt + 1 < NK) {
      commitB(buf ^ 1, rb);
      asm volatile("s_wait_asynccnt 0" ::: "memory");  // my DMA done
      __syncthreads();                                  // all DMAs visible
      buf ^= 1;
    }
  }

#pragma unroll
  for (int fm = 0; fm < 4; ++fm)
#pragma unroll
    for (int fn = 0; fn < 4; ++fn) {
      const int col = wn * 64 + fn * 16 + lr;
      const float bv = biasg[col];
#pragma unroll
      for (int r = 0; r < 8; ++r) {
        const int row = wm * 64 + fm * 16 + half + r;
        Og[(size_t)row * D_ + col] = acc[fm][fn][r] + bv;
      }
    }
}

extern "C" void kernel_launch(void* const* d_in, const int* in_sizes, int n_in,
                              void* d_out, int out_size, void* d_ws, size_t ws_size,
                              hipStream_t stream) {
  const float* x  = (const float*)d_in[0];
  const float* w1 = (const float*)d_in[1];
  const float* b1 = (const float*)d_in[2];
  const float* w2 = (const float*)d_in[3];
  const float* b2 = (const float*)d_in[4];
  float* out = (float*)d_out;
  ushort_t* ybuf = (ushort_t*)d_ws;  // E*T*H bf16 = 128 MiB intermediate

  dim3 block(256);
  dim3 g1(H_ / K1_BN, T_ / K1_BM, E_);  // 16 x 16 x 8
  dim3 g2(D_ / K2_BN, T_ / K2_BM, E_);  //  8 x  8 x 8
  moe_fc1_relu<<<g1, block, 0, stream>>>(x, w1, b1, ybuf);
  moe_fc2<<<g2, block, 0, stream>>>(ybuf, w2, b2, out);
}
